// TableTransform_1159641170471
// MI455X (gfx1250) — matrix operand order repaired and checked by reading, not compiled, not run on hardware
//
#include <hip/hip_runtime.h>
#include <hip/hip_bf16.h>
#include <math.h>

// ---------------------------------------------------------------------------
// Problem constants (fixed by the reference)
// ---------------------------------------------------------------------------
#define NT   8000
#define NC   80000
#define NE   80000
#define HID  256
#define GATH 8
#define T2C  32
#define AGGH 64
#define PH   16
#define NPRED 2
#define STAT 40
#define TFEAT 8
#define DENSE 64
#define DIN_ALL (TFEAT + HID + DENSE)   // 328
#define DIN_FIN (HID * NPRED + TFEAT)   // 520

typedef __attribute__((ext_vector_type(16))) _Float16 v16h;
typedef __attribute__((ext_vector_type(8)))  float    v8f;

// ---------------------------------------------------------------------------
// One 64x64 WMMA macro-step from staged LDS tiles (ISA 7.12.2 layouts).
// ---------------------------------------------------------------------------
__device__ __forceinline__ void wmma_tile_step(_Float16 (*As)[40], _Float16 (*Bs)[40],
                                               int wm, int wn, int mlo, int kh,
                                               v8f& acc0, v8f& acc1)
{
    v16h a0, a1, bf;
    const int r0 = wm * 32 + mlo;
    const int r1 = r0 + 16;
    const int nb = wn * 16 + mlo;
    #pragma unroll
    for (int v = 0; v < 8; ++v) {
        const int kb  = 2 * v + (v >= 4 ? 8 : 0) + 8 * kh;  // A K index
        a0[2 * v]     = As[r0][kb];
        a0[2 * v + 1] = As[r0][kb + 1];
        a1[2 * v]     = As[r1][kb];
        a1[2 * v + 1] = As[r1][kb + 1];
        const int kbb = 2 * v + (v >= 4 ? 8 : 0) + 8 * kh;  // B K index
        bf[2 * v]     = Bs[nb][kbb];
        bf[2 * v + 1] = Bs[nb][kbb + 1];
    }
    acc0 = __builtin_amdgcn_wmma_f32_16x16x32_f16(false, a0, false, bf,
                                                  (short)0, acc0, false, false);
    acc1 = __builtin_amdgcn_wmma_f32_16x16x32_f16(false, a1, false, bf,
                                                  (short)0, acc1, false, false);
}

// ---------------------------------------------------------------------------
// Generic tiled WMMA GEMM: C[M,N] = act(A[M,K]) @ B[K,N] (+ bias[N])
// Preconditions guaranteed by the launcher:
//   - M % 64 == 0 (no row guards)
//   - K % 4 == 0, N % 4 == 0, 16B-aligned bases (vectorized staging)
// Block = 256 threads = 8 waves; tile 64(M) x 64(N); K step 32.
// ---------------------------------------------------------------------------
__global__ __launch_bounds__(256)
void gemm_wmma_f16(const float* __restrict__ A, const float* __restrict__ B,
                   float* __restrict__ C, const float* __restrict__ bias,
                   int M, int N, int K, int act_in)
{
    __shared__ __align__(16) _Float16 As[64][40];   // [m][k]
    __shared__ __align__(16) _Float16 Bs[64][40];   // [n][k]  (transposed on load)

    const int tid  = threadIdx.x;
    const int lane = tid & 31;
    const int wave = tid >> 5;
    const int wm   = wave >> 2;   // 0..1
    const int wn   = wave & 3;    // 0..3
    const int bm   = blockIdx.y * 64;
    const int bn   = blockIdx.x * 64;
    const bool nEdge = (bn + 64 > N);   // block-uniform

    const int mlo = lane & 15;
    const int kh  = lane >> 4;

    // Staging geometry: A -> thread owns rows {ar, ar+32} x 4 cols at ac4
    //                   B -> thread owns k-rows {br, br+16} x 4 cols at bc4
    const int ar  = tid >> 3;          // 0..31
    const int ac4 = (tid & 7) * 4;     // 0..28
    const int br  = tid >> 4;          // 0..15
    const int bc4 = (tid & 15) * 4;    // 0..60

    v8f acc0 = {};
    v8f acc1 = {};

    const int kFull = K & ~31;

    for (int kk = 0; kk < kFull; kk += 32) {
        // ---- stage A tile 64x32 (two float4 loads per thread, no guards) ----
        #pragma unroll
        for (int rr = 0; rr < 2; ++rr) {
            const int r = ar + rr * 32;
            float4 v = *(const float4*)(A + (size_t)(bm + r) * K + kk + ac4);
            if (act_in) {
                v.x = v.x > 0.0f ? v.x : 0.01f * v.x;
                v.y = v.y > 0.0f ? v.y : 0.01f * v.y;
                v.z = v.z > 0.0f ? v.z : 0.01f * v.z;
                v.w = v.w > 0.0f ? v.w : 0.01f * v.w;
            }
            union { _Float16 h[4]; uint2 u; } pk;
            pk.h[0] = (_Float16)v.x; pk.h[1] = (_Float16)v.y;
            pk.h[2] = (_Float16)v.z; pk.h[3] = (_Float16)v.w;
            *(uint2*)&As[r][ac4] = pk.u;
        }
        // ---- stage B tile 32x64 transposed into Bs[n][k] ----
        if (!nEdge) {
            #pragma unroll
            for (int rr = 0; rr < 2; ++rr) {
                const int kr = br + rr * 16;
                const float4 v = *(const float4*)(B + (size_t)(kk + kr) * N + bn + bc4);
                Bs[bc4 + 0][kr] = (_Float16)v.x;
                Bs[bc4 + 1][kr] = (_Float16)v.y;
                Bs[bc4 + 2][kr] = (_Float16)v.z;
                Bs[bc4 + 3][kr] = (_Float16)v.w;
            }
        } else {
            #pragma unroll
            for (int rr = 0; rr < 2; ++rr) {
                const int kr = br + rr * 16;
                #pragma unroll
                for (int j = 0; j < 4; ++j) {
                    const int gn = bn + bc4 + j;
                    float f = (gn < N) ? B[(size_t)(kk + kr) * N + gn] : 0.0f;
                    Bs[bc4 + j][kr] = (_Float16)f;
                }
            }
        }
        // Prefetch next K-chunk (speculative; dropped if OOB per ISA 10.5)
        __builtin_prefetch((const void*)(A + (size_t)(bm + ar) * K + kk + 32 + ac4), 0, 0);
        __builtin_prefetch((const void*)(B + (size_t)(kk + 32 + br) * N + bn + bc4), 0, 0);

        __syncthreads();
        wmma_tile_step(As, Bs, wm, wn, mlo, kh, acc0, acc1);
        __syncthreads();
    }

    // ---- K remainder chunk (runs at most once; guarded scalar staging) ----
    if (kFull < K) {
        const int kk = kFull;
        for (int i = tid; i < 64 * 32; i += 256) {
            const int r = i >> 5, c = i & 31;
            float v = 0.0f;
            if (kk + c < K) v = A[(size_t)(bm + r) * K + kk + c];
            if (act_in) v = v > 0.0f ? v : 0.01f * v;
            As[r][c] = (_Float16)v;
        }
        for (int i = tid; i < 32 * 64; i += 256) {
            const int r = i >> 6, c = i & 63;  // r = k, c = n
            float v = 0.0f;
            if (kk + r < K && bn + c < N) v = B[(size_t)(kk + r) * N + bn + c];
            Bs[c][r] = (_Float16)v;
        }
        __syncthreads();
        wmma_tile_step(As, Bs, wm, wn, mlo, kh, acc0, acc1);
        __syncthreads();
    }

    // ---- store D: f32 16x16 layout (lane n = lane&15; VGPR v -> m = v+8*kh) ----
    const int col = bn + wn * 16 + mlo;
    if (nEdge && col >= N) return;
    float bv = 0.0f;
    if (bias != nullptr) bv = bias[col];
    #pragma unroll
    for (int v = 0; v < 8; ++v) {
        const int m0 = bm + wm * 32 + v + 8 * kh;
        C[(size_t)m0 * N + col] = acc0[v] + bv;
        C[(size_t)(m0 + 16) * N + col] = acc1[v] + bv;
    }
}

// ---------------------------------------------------------------------------
// LoRA weight folding: Weff[K,N] = W[N,K]^T + 2 * A[8,K]^T @ B[N,8]^T
// ---------------------------------------------------------------------------
__global__ void fold_lora_T(const float* __restrict__ W, const float* __restrict__ A,
                            const float* __restrict__ Bm, float* __restrict__ out,
                            int N, int Kd)
{
    int idx = blockIdx.x * blockDim.x + threadIdx.x;
    if (idx >= N * Kd) return;
    int k = idx / N, n = idx % N;
    float s = 0.0f;
    #pragma unroll
    for (int r = 0; r < 8; ++r) s += A[r * Kd + k] * Bm[n * 8 + r];
    out[(size_t)k * N + n] = W[(size_t)n * Kd + k] + 2.0f * s;
}

// Plain transpose: out[Kd,N] = W[N,Kd]^T
__global__ void transpose_T(const float* __restrict__ W, float* __restrict__ out,
                            int N, int Kd)
{
    int idx = blockIdx.x * blockDim.x + threadIdx.x;
    if (idx >= N * Kd) return;
    int k = idx / N, n = idx % N;
    out[(size_t)k * N + n] = W[(size_t)n * Kd + k];
}

// Cluster LoRA embedding fold: WclE[33,256] = W_cl + 2*A_cl@B_cl
__global__ void fold_cl_k(const float* __restrict__ Wcl, const float* __restrict__ Acl,
                          const float* __restrict__ Bcl, float* __restrict__ out)
{
    int idx = blockIdx.x * blockDim.x + threadIdx.x;
    if (idx >= 33 * 256) return;
    int c = idx >> 8, d = idx & 255;
    float s = 0.0f;
    #pragma unroll
    for (int r = 0; r < 8; ++r) s += Acl[c * 8 + r] * Bcl[r * 256 + d];
    out[idx] = Wcl[idx] + 2.0f * s;
}

// Gate fold: WgE[p][k][d] = Wg[p][d][k] + 2*sum_r Ag[p][r][k]*Bg[p][d][r]
__global__ void fold_g_k(const float* __restrict__ Wg, const float* __restrict__ Ag,
                         const float* __restrict__ Bg, float* __restrict__ out)
{
    int idx = blockIdx.x * blockDim.x + threadIdx.x;
    if (idx >= NPRED * PH * HID) return;
    int p = idx / (PH * HID);
    int k = (idx >> 8) & (PH - 1);
    int d = idx & 255;
    float s = 0.0f;
    #pragma unroll
    for (int r = 0; r < 8; ++r) s += Ag[p * 128 + r * 16 + k] * Bg[p * 2048 + d * 8 + r];
    out[idx] = Wg[p * 4096 + d * 16 + k] + 2.0f * s;
}

// ---------------------------------------------------------------------------
// Build tf (sqrt transform) and x = concat([tf, cl_emb + t_emb, dense])
// ---------------------------------------------------------------------------
__global__ __launch_bounds__(128)
void build_x_k(const float* __restrict__ tfeat, const float* __restrict__ dense,
               const int* __restrict__ cid, const int* __restrict__ tidv,
               const float* __restrict__ WclE, const float* __restrict__ Wtab,
               const float* __restrict__ WtabF, float* __restrict__ tf,
               float* __restrict__ x)
{
    const int t  = blockIdx.x;
    const int th = threadIdx.x;
    const int c  = cid[t];
    const int ti = tidv[t];
    if (th < TFEAT) {
        float v = tfeat[t * TFEAT + th];
        float s = (v >= 0.0f ? 1.0f : -1.0f) * sqrtf(fabsf(v));
        tf[t * TFEAT + th] = s;
        x[(size_t)t * DIN_ALL + th] = s;
    }
    for (int d = th; d < HID; d += 128)
        x[(size_t)t * DIN_ALL + TFEAT + d] =
            WclE[c * HID + d] + Wtab[ti * HID + d] + WtabF[ti * HID + d];
    for (int d = th; d < DENSE; d += 128)
        x[(size_t)t * DIN_ALL + TFEAT + HID + d] = dense[t * DENSE + d];
}

// el/er: out[n,h] = sum_d f[n,h,d] * attn[h,d]
__global__ void rowdot8_k(const float* __restrict__ f, const float* __restrict__ attn,
                          float* __restrict__ out, int Nrows)
{
    int idx = blockIdx.x * blockDim.x + threadIdx.x;
    if (idx >= Nrows * GATH) return;
    int n = idx >> 3, h = idx & 7;
    const float* fr = f + (size_t)n * (GATH * HID) + h * HID;
    const float* ar = attn + h * HID;
    float s = 0.0f;
    for (int d = 0; d < HID; d += 4) {
        const float4 a = *(const float4*)(fr + d);
        const float4 b = *(const float4*)(ar + d);
        s += a.x * b.x + a.y * b.y + a.z * b.z + a.w * b.w;
    }
    out[idx] = s;
}

__global__ void fill_k(float* __restrict__ p, float v, int n)
{
    int i = blockIdx.x * blockDim.x + threadIdx.x;
    if (i < n) p[i] = v;
}

__device__ inline void atomicMaxF(float* addr, float v)
{
    if (v >= 0.0f) atomicMax((int*)addr, __float_as_int(v));
    else           atomicMin((unsigned int*)addr, (unsigned int)__float_as_int(v));
}

__global__ void edge_max_k(const int* __restrict__ es, const int* __restrict__ ed,
                           const float* __restrict__ el, const float* __restrict__ er,
                           float* __restrict__ m)
{
    int idx = blockIdx.x * blockDim.x + threadIdx.x;
    if (idx >= NE * GATH) return;
    int e = idx >> 3, h = idx & 7;
    float v = el[es[e] * GATH + h] + er[ed[e] * GATH + h];
    v = v > 0.0f ? v : 0.2f * v;
    atomicMaxF(&m[ed[e] * GATH + h], v);
}

__global__ void edge_exp_k(const int* __restrict__ es, const int* __restrict__ ed,
                           const float* __restrict__ el, const float* __restrict__ er,
                           const float* __restrict__ m, float* __restrict__ ee,
                           float* __restrict__ z)
{
    int idx = blockIdx.x * blockDim.x + threadIdx.x;
    if (idx >= NE * GATH) return;
    int e = idx >> 3, h = idx & 7;
    int d = ed[e];
    float v = el[es[e] * GATH + h] + er[d * GATH + h];
    v = v > 0.0f ? v : 0.2f * v;
    float x = expf(v - m[d * GATH + h]);
    ee[idx] = x;
    atomicAdd(&z[d * GATH + h], x);
}

// Scatter messages: rst[dst,h,:] += alpha * fs[src,h,:]
__global__ __launch_bounds__(256)
void edge_msg_k(const int* __restrict__ es, const int* __restrict__ ed,
                const float* __restrict__ ee, const float* __restrict__ z,
                const float* __restrict__ fs, float* __restrict__ rst)
{
    const int e = blockIdx.x;
    __shared__ float al[GATH];
    const int s = es[e], d = ed[e];
    if (threadIdx.x < GATH)
        al[threadIdx.x] = ee[e * GATH + threadIdx.x] / (z[d * GATH + threadIdx.x] + 1e-9f);
    __syncthreads();
    const int t = threadIdx.x;
    #pragma unroll
    for (int h = 0; h < GATH; ++h)
        atomicAdd(&rst[(size_t)d * 2048 + h * HID + t],
                  al[h] * fs[(size_t)s * 2048 + h * HID + t]);
}

__global__ void mean_heads_k(const float* __restrict__ rst, const float* __restrict__ res,
                             float* __restrict__ tn)
{
    int idx = blockIdx.x * blockDim.x + threadIdx.x;
    if (idx >= NT * HID) return;
    int t = idx >> 8, d = idx & 255;
    float s = 0.0f;
    #pragma unroll
    for (int h = 0; h < GATH; ++h)
        s += rst[(size_t)t * 2048 + h * HID + d] + res[(size_t)t * 2048 + h * HID + d];
    tn[idx] = s * 0.125f;
}

// edge_out[e,k] = sum_j cs[src,j] * tc[dst,j,k]
__global__ __launch_bounds__(64)
void edge_dot_k(const int* __restrict__ es, const int* __restrict__ ed,
                const float* __restrict__ cs, const float* __restrict__ tc,
                float* __restrict__ eout)
{
    const int e = blockIdx.x;
    __shared__ float css[T2C];
    const int s = es[e], d = ed[e];
    if (threadIdx.x < T2C) css[threadIdx.x] = cs[(size_t)s * T2C + threadIdx.x];
    __syncthreads();
    const int k = threadIdx.x;
    float acc = 0.0f;
    #pragma unroll
    for (int j = 0; j < T2C; ++j)
        acc += css[j] * tc[(size_t)d * 2048 + j * AGGH + k];
    eout[(size_t)e * AGGH + k] = acc;
}

// ph[e, p*16+k] *= sqrt_tf(pred_feat[src,p]) * mask[src,p]
__global__ void scale_ph_k(const int* __restrict__ es, const float* __restrict__ pf,
                           const int* __restrict__ pm, float* __restrict__ ph)
{
    int idx = blockIdx.x * blockDim.x + threadIdx.x;
    if (idx >= NE * 32) return;
    int e = idx >> 5, j = idx & 31, p = j >> 4;
    int s = es[e];
    float v = pf[s * NPRED + p];
    float pv = (v >= 0.0f ? 1.0f : -1.0f) * sqrtf(fabsf(v)) * (float)pm[s * NPRED + p];
    ph[idx] *= pv;
}

// pe[e,p,d] = sum_k ph[e,p,k]*WgE[p,k,d]; scatter-add into agg[dst], count edges
__global__ __launch_bounds__(256)
void edge_pe_k(const int* __restrict__ ed, const float* __restrict__ ph,
               const float* __restrict__ WgE, float* __restrict__ agg,
               float* __restrict__ cnt)
{
    const int e = blockIdx.x;
    __shared__ float phs[32];
    const int d = ed[e];
    if (threadIdx.x < 32) phs[threadIdx.x] = ph[(size_t)e * 32 + threadIdx.x];
    __syncthreads();
    const int t = threadIdx.x;
    #pragma unroll
    for (int p = 0; p < NPRED; ++p) {
        float acc = 0.0f;
        #pragma unroll
        for (int k = 0; k < PH; ++k)
            acc += phs[p * PH + k] * WgE[p * 4096 + k * HID + t];
        atomicAdd(&agg[(size_t)d * (NPRED * HID) + p * HID + t], acc);
    }
    if (threadIdx.x == 0) atomicAdd(&cnt[d], 1.0f);
}

// xfin = concat([agg / max(cnt,1), tf])
__global__ __launch_bounds__(256)
void fin_x_k(const float* __restrict__ agg, const float* __restrict__ cnt,
             const float* __restrict__ tf, float* __restrict__ xf)
{
    const int t = blockIdx.x;
    float c = cnt[t];
    c = c > 1.0f ? c : 1.0f;
    const float inv = 1.0f / c;
    const int th = threadIdx.x;
    for (int d = th; d < NPRED * HID; d += 256)
        xf[(size_t)t * DIN_FIN + d] = agg[(size_t)t * (NPRED * HID) + d] * inv;
    if (th < TFEAT) xf[(size_t)t * DIN_FIN + NPRED * HID + th] = tf[t * TFEAT + th];
}

// ---------------------------------------------------------------------------
// Host launcher
// ---------------------------------------------------------------------------
extern "C" void kernel_launch(void* const* d_in, const int* in_sizes, int n_in,
                              void* d_out, int out_size, void* d_ws, size_t ws_size,
                              hipStream_t stream)
{
    (void)in_sizes; (void)n_in; (void)out_size; (void)ws_size;

    const float* table_features = (const float*)d_in[0];
    const float* table_dense    = (const float*)d_in[1];
    const float* column_stat    = (const float*)d_in[2];
    const float* pred_feat      = (const float*)d_in[3];
    const float* W_cl   = (const float*)d_in[4];
    const float* A_cl   = (const float*)d_in[5];
    const float* B_cl   = (const float*)d_in[6];
    const float* W_tab  = (const float*)d_in[7];
    const float* W_tabF = (const float*)d_in[8];
    const float* W_all  = (const float*)d_in[9];
    const float* A_all  = (const float*)d_in[10];
    const float* B_all  = (const float*)d_in[11];
    const float* W_src  = (const float*)d_in[12];
    const float* W_dst  = (const float*)d_in[13];
    const float* attn_l = (const float*)d_in[14];
    const float* attn_r = (const float*)d_in[15];
    const float* W_res  = (const float*)d_in[16];
    const float* W_cs   = (const float*)d_in[17];
    const float* A_cs   = (const float*)d_in[18];
    const float* B_cs   = (const float*)d_in[19];
    const float* W_t2c  = (const float*)d_in[20];
    const float* A_t2c  = (const float*)d_in[21];
    const float* B_t2c  = (const float*)d_in[22];
    const float* b_t2c  = (const float*)d_in[23];
    const float* W_ph   = (const float*)d_in[24];
    const float* A_ph   = (const float*)d_in[25];
    const float* B_ph   = (const float*)d_in[26];
    const float* b_ph   = (const float*)d_in[27];
    const float* Wg     = (const float*)d_in[28];
    const float* Ag     = (const float*)d_in[29];
    const float* Bg     = (const float*)d_in[30];
    const float* W_fin  = (const float*)d_in[31];
    const float* A_fin  = (const float*)d_in[32];
    const float* B_fin  = (const float*)d_in[33];
    const int* cluster_id = (const int*)d_in[34];
    const int* table_id   = (const int*)d_in[35];
    const int* pred_mask  = (const int*)d_in[36];
    const int* edge_src   = (const int*)d_in[37];
    const int* edge_dst   = (const int*)d_in[38];

    float* ws = (float*)d_ws;
    size_t off = 0;
    auto alloc = [&](size_t n) { float* p = ws + off; off += n; return p; };

    float* tf    = alloc((size_t)NT * TFEAT);
    float* x     = alloc((size_t)NT * DIN_ALL);
    float* temb  = alloc((size_t)NT * HID);
    float* WclE  = alloc(33 * 256);
    float* WallE = alloc((size_t)DIN_ALL * HID);
    float* WsrcT = alloc((size_t)STAT * 2048);
    float* WdstT = alloc((size_t)HID * 2048);
    float* WresT = alloc((size_t)HID * 2048);
    float* WcsE  = alloc((size_t)STAT * T2C);
    float* Wt2cE = alloc((size_t)HID * 2048);
    float* WphE  = alloc((size_t)AGGH * 32);
    float* WgE   = alloc((size_t)NPRED * PH * HID);
    float* WfinE = alloc((size_t)DIN_FIN * HID);
    float* fs    = alloc((size_t)NC * 2048);
    float* fd    = alloc((size_t)NT * 2048);
    float* resb  = alloc((size_t)NT * 2048);
    float* el    = alloc((size_t)NC * GATH);
    float* er    = alloc((size_t)NT * GATH);
    float* mmax  = alloc((size_t)NT * GATH);
    float* zsum  = alloc((size_t)NT * GATH);
    float* eew   = alloc((size_t)NE * GATH);
    float* rst   = alloc((size_t)NT * 2048);
    float* tnew  = alloc((size_t)NT * HID);
    float* csb   = alloc((size_t)NC * T2C);
    float* tcb   = alloc((size_t)NT * 2048);
    float* eout  = alloc((size_t)NE * AGGH);
    float* phb   = alloc((size_t)NE * 32);
    float* agg   = alloc((size_t)NT * NPRED * HID);
    float* cnt   = alloc((size_t)NT);
    float* xfin  = alloc((size_t)NT * DIN_FIN);

    const int TPB = 256;
    auto g1 = [](size_t n) { return (unsigned)((n + 255) / 256); };
    auto gemm = [&](const float* A, const float* B, float* C, const float* bias,
                    int M, int N, int K, int act) {
        dim3 grid((unsigned)((N + 63) / 64), (unsigned)((M + 63) / 64));
        gemm_wmma_f16<<<grid, dim3(256), 0, stream>>>(A, B, C, bias, M, N, K, act);
    };

    // ---- fold LoRA weights into effective GEMM weights ----
    fold_cl_k  <<<g1(33 * 256), TPB, 0, stream>>>(W_cl, A_cl, B_cl, WclE);
    fold_lora_T<<<g1((size_t)DIN_ALL * HID), TPB, 0, stream>>>(W_all, A_all, B_all, WallE, HID, DIN_ALL);
    transpose_T<<<g1((size_t)2048 * STAT), TPB, 0, stream>>>(W_src, WsrcT, 2048, STAT);
    transpose_T<<<g1((size_t)2048 * HID), TPB, 0, stream>>>(W_dst, WdstT, 2048, HID);
    transpose_T<<<g1((size_t)2048 * HID), TPB, 0, stream>>>(W_res, WresT, 2048, HID);
    fold_lora_T<<<g1((size_t)STAT * T2C), TPB, 0, stream>>>(W_cs, A_cs, B_cs, WcsE, T2C, STAT);
    fold_lora_T<<<g1((size_t)HID * 2048), TPB, 0, stream>>>(W_t2c, A_t2c, B_t2c, Wt2cE, 2048, HID);
    fold_lora_T<<<g1((size_t)AGGH * 32), TPB, 0, stream>>>(W_ph, A_ph, B_ph, WphE, 32, AGGH);
    fold_g_k   <<<g1((size_t)NPRED * PH * HID), TPB, 0, stream>>>(Wg, Ag, Bg, WgE);
    fold_lora_T<<<g1((size_t)DIN_FIN * HID), TPB, 0, stream>>>(W_fin, A_fin, B_fin, WfinE, HID, DIN_FIN);

    // ---- table embedding ----
    build_x_k<<<NT, 128, 0, stream>>>(table_features, table_dense, cluster_id,
                                      table_id, WclE, W_tab, W_tabF, tf, x);
    gemm(x, WallE, temb, nullptr, NT, HID, DIN_ALL, 0);

    // ---- GAT feature projections (WMMA GEMMs) ----
    gemm(column_stat, WsrcT, fs,   nullptr, NC, 2048, STAT, 0);
    gemm(temb,        WdstT, fd,   nullptr, NT, 2048, HID,  0);
    gemm(temb,        WresT, resb, nullptr, NT, 2048, HID,  0);
    rowdot8_k<<<g1((size_t)NC * GATH), TPB, 0, stream>>>(fs, attn_l, el, NC);
    rowdot8_k<<<g1((size_t)NT * GATH), TPB, 0, stream>>>(fd, attn_r, er, NT);

    // ---- segment softmax + message aggregation ----
    fill_k<<<g1((size_t)NT * GATH), TPB, 0, stream>>>(mmax, -__builtin_inff(), NT * GATH);
    hipMemsetAsync(zsum, 0, (size_t)NT * GATH * sizeof(float), stream);
    hipMemsetAsync(rst,  0, (size_t)NT * 2048 * sizeof(float), stream);
    hipMemsetAsync(agg,  0, (size_t)NT * NPRED * HID * sizeof(float), stream);
    hipMemsetAsync(cnt,  0, (size_t)NT * sizeof(float), stream);

    edge_max_k<<<g1((size_t)NE * GATH), TPB, 0, stream>>>(edge_src, edge_dst, el, er, mmax);
    edge_exp_k<<<g1((size_t)NE * GATH), TPB, 0, stream>>>(edge_src, edge_dst, el, er, mmax, eew, zsum);
    edge_msg_k<<<NE, 256, 0, stream>>>(edge_src, edge_dst, eew, zsum, fs, rst);
    mean_heads_k<<<g1((size_t)NT * HID), TPB, 0, stream>>>(rst, resb, tnew);

    // ---- per-edge einsum chain ----
    gemm(column_stat, WcsE, csb, nullptr, NC, T2C, STAT, 0);
    gemm(tnew, Wt2cE, tcb, b_t2c, NT, 2048, HID, 0);
    edge_dot_k<<<NE, 64, 0, stream>>>(edge_src, edge_dst, csb, tcb, eout);
    gemm(eout, WphE, phb, b_ph, NE, 32, AGGH, 1 /* leaky 0.01 on input */);
    scale_ph_k<<<g1((size_t)NE * 32), TPB, 0, stream>>>(edge_src, pred_feat, pred_mask, phb);
    edge_pe_k<<<NE, 256, 0, stream>>>(edge_dst, phb, WgE, agg, cnt);

    // ---- final LoRA linear ----
    fin_x_k<<<NT, 256, 0, stream>>>(agg, cnt, tf, xfin);
    gemm(xfin, WfinE, (float*)d_out, nullptr, NT, HID, DIN_FIN, 0);
}
